// SAModule_88459146428521
// MI455X (gfx1250) — matrix operand order repaired and checked by reading, not compiled, not run on hardware
//
#include <hip/hip_runtime.h>
#include <hip/hip_fp16.h>
#include <math.h>

typedef __attribute__((ext_vector_type(16))) _Float16 v16h;
typedef __attribute__((ext_vector_type(8)))  _Float16 v8h;
typedef __attribute__((ext_vector_type(8)))  float    v8f;

#define BB    8
#define NN    4096
#define NINF  64
#define NOUTF 128
#define KK    32
#define MM    1024
#define DIMF  68
#define DIMP  96          // padded DIM (3 k-chunks of 32, 6 col tiles of 16)
#define R2C   0.0225f     // R*R, R=0.15
#define NCTR  (BB*MM)     // 8192 centers

// packed fragment-layout weight sizes (halves)
#define W12P_ELEMS (3*6*32*16)   // 9216  : 96x96 -> (kc,ct) tiles, per-lane 16 halves
#define W3P_ELEMS  (3*8*32*16)   // 12288 : 96x128

// ---------------- workspace layout (bytes) ----------------
static constexpr size_t OFF_W1P  = 0;
static constexpr size_t OFF_W2P  = OFF_W1P  + (size_t)W12P_ELEMS*2;   // 18432
static constexpr size_t OFF_W3P  = OFF_W2P  + (size_t)W12P_ELEMS*2;   // 36864
static constexpr size_t OFF_NRMS = OFF_W3P  + (size_t)W3P_ELEMS*2;    // 61440
static constexpr size_t OFF_NBR  = OFF_NRMS + (size_t)NCTR*3*4;       // 159744
static constexpr size_t OFF_AGG  = OFF_NBR  + (size_t)NCTR*KK*4;      // 1208320 (16B aligned)

union F16x16 { v16h v; _Float16 e[16]; };
union F32x8  { v8f  v; float    e[8];  };

// ---------------- weight conversion: f32 -> fragment-packed f16 ----------------
// Packed layout: index ((kc*NT + ct)*32 + lane)*16 + h holds
// W[kc*32 + 8*(lane>>4) + (h&7) + 16*(h>>3)][ct*16 + (lane&15)]  (zero outside real dims).
__global__ void prep_weights_kernel(const float* __restrict__ W1,
                                    const float* __restrict__ W2,
                                    const float* __restrict__ W3,
                                    _Float16* __restrict__ W1p,
                                    _Float16* __restrict__ W2p,
                                    _Float16* __restrict__ W3p) {
  int i = blockIdx.x * blockDim.x + threadIdx.x;
  if (i < W12P_ELEMS) {                 // 6 col-tiles of 96x96 layers
    int h = i & 15, lane = (i >> 4) & 31, t = i >> 9;   // t = kc*6+ct
    int kc = t / 6, ct = t % 6;
    int kk = kc * 32 + 8 * (lane >> 4) + (h & 7) + 16 * (h >> 3);
    int nn = ct * 16 + (lane & 15);
    bool in1 = (kk < DIMF) && (nn < DIMF);
    W1p[i] = (_Float16)(in1 ? W1[kk * DIMF + nn] : 0.f);
    W2p[i] = (_Float16)(in1 ? W2[kk * DIMF + nn] : 0.f);
  }
  if (i < W3P_ELEMS) {                  // 8 col-tiles of 96x128
    int h = i & 15, lane = (i >> 4) & 31, t = i >> 9;   // t = kc*8+ct
    int kc = t / 8, ct = t % 8;
    int kk = kc * 32 + 8 * (lane >> 4) + (h & 7) + 16 * (h >> 3);
    int nn = ct * 16 + (lane & 15);
    W3p[i] = (_Float16)((kk < DIMF) ? W3[kk * NOUTF + nn] : 0.f);
  }
}

// ---------------- farthest point sampling: one block per cloud ----------------
__global__ void fps_kernel(const float* __restrict__ pos,
                           const float* __restrict__ norm,
                           float* __restrict__ pos_s,
                           float* __restrict__ norm_s,
                           int* __restrict__ idx_out) {
  const int b = blockIdx.x;
  const int tid = threadIdx.x;           // 256 threads
  const float* posb = pos + (size_t)b * NN * 3;
  const float* nrmb = norm + (size_t)b * NN * 3;

  float dloc[NN / 256];
  float px[NN / 256], py[NN / 256], pz[NN / 256];
#pragma unroll
  for (int j = 0; j < NN / 256; ++j) {
    int p = tid + 256 * j;
    dloc[j] = __builtin_inff();
    px[j] = posb[p * 3 + 0];
    py[j] = posb[p * 3 + 1];
    pz[j] = posb[p * 3 + 2];
  }

  __shared__ float rv[256];
  __shared__ int   ri[256];
  __shared__ int   sLast;

  int last = 0;
  for (int i = 0; i < MM; ++i) {
    if (tid == 0) {
      idx_out[b * MM + i] = last;
      pos_s[((size_t)b * MM + i) * 3 + 0] = posb[last * 3 + 0];
      pos_s[((size_t)b * MM + i) * 3 + 1] = posb[last * 3 + 1];
      pos_s[((size_t)b * MM + i) * 3 + 2] = posb[last * 3 + 2];
      norm_s[((size_t)b * MM + i) * 3 + 0] = nrmb[last * 3 + 0];
      norm_s[((size_t)b * MM + i) * 3 + 1] = nrmb[last * 3 + 1];
      norm_s[((size_t)b * MM + i) * 3 + 2] = nrmb[last * 3 + 2];
    }
    const float cx = posb[last * 3 + 0];
    const float cy = posb[last * 3 + 1];
    const float cz = posb[last * 3 + 2];
    float lv = -1.f;
    int   li = 0x7fffffff;
#pragma unroll
    for (int j = 0; j < NN / 256; ++j) {
      float dx = px[j] - cx, dy = py[j] - cy, dz = pz[j] - cz;
      float d = dx * dx + dy * dy + dz * dz;
      float nd = fminf(dloc[j], d);
      dloc[j] = nd;
      int p = tid + 256 * j;
      if (nd > lv || (nd == lv && p < li)) { lv = nd; li = p; }
    }
    rv[tid] = lv; ri[tid] = li;
    __syncthreads();
    for (int s = 128; s > 0; s >>= 1) {
      if (tid < s) {
        if (rv[tid + s] > rv[tid] ||
            (rv[tid + s] == rv[tid] && ri[tid + s] < ri[tid])) {
          rv[tid] = rv[tid + s]; ri[tid] = ri[tid + s];
        }
      }
      __syncthreads();
    }
    if (tid == 0) sLast = ri[0];
    __syncthreads();
    last = sLast;
    __syncthreads();
  }
}

// ---------------- K nearest within radius: one block per center ----------------
__global__ void nbr_kernel(const float* __restrict__ pos,
                           const float* __restrict__ pos_s,
                           int* __restrict__ nbr) {
  const int c = blockIdx.x;              // 0..8191
  const int b = c / MM;
  const int tid = threadIdx.x;           // 256 threads
  __shared__ float d2s[NN];
  __shared__ float rv[256];
  __shared__ int   ri[256];

  const float cx = pos_s[(size_t)c * 3 + 0];
  const float cy = pos_s[(size_t)c * 3 + 1];
  const float cz = pos_s[(size_t)c * 3 + 2];
  const float* posb = pos + (size_t)b * NN * 3;

  for (int p = tid; p < NN; p += 256) {
    float dx = posb[p * 3 + 0] - cx;
    float dy = posb[p * 3 + 1] - cy;
    float dz = posb[p * 3 + 2] - cz;
    float d = dx * dx + dy * dy + dz * dz;
    d2s[p] = (d <= R2C) ? d : __builtin_inff();
  }
  __syncthreads();

  for (int k = 0; k < KK; ++k) {
    float lv = __builtin_inff();
    int   li = 0x7fffffff;
    for (int p = tid; p < NN; p += 256) {
      float d = d2s[p];
      if (d < lv || (d == lv && p < li)) { lv = d; li = p; }
    }
    rv[tid] = lv; ri[tid] = li;
    __syncthreads();
    for (int s = 128; s > 0; s >>= 1) {
      if (tid < s) {
        if (rv[tid + s] < rv[tid] ||
            (rv[tid + s] == rv[tid] && ri[tid + s] < ri[tid])) {
          rv[tid] = rv[tid + s]; ri[tid] = ri[tid + s];
        }
      }
      __syncthreads();
    }
    if (tid == 0) {
      float v = rv[0]; int w = ri[0];
      bool ok = (v < __builtin_inff()) && (w >= 0) && (w < NN);
      nbr[(size_t)c * KK + k] = ok ? w : -1;
      if (ok) d2s[w] = __builtin_inff();
    }
    __syncthreads();
  }
}

// ---------------- vectorized WMMA fragment loads ----------------
// A 16x32 f16 (row-major [row][DIMP] tile): lane's halves 0-7 = K kc*32+8*ksel+(0..7),
// halves 8-15 = +16. Two 16B chunks -> two b128 loads.
__device__ __forceinline__ v16h load_a_frag(const _Float16* rowp, int kc, int ksel) {
  const v8h lo = *(const v8h*)(rowp + kc * 32 + 8 * ksel);
  const v8h hi = *(const v8h*)(rowp + kc * 32 + 16 + 8 * ksel);
  return __builtin_shufflevector(lo, hi, 0, 1, 2, 3, 4, 5, 6, 7,
                                 8, 9, 10, 11, 12, 13, 14, 15);
}
// B fragment from fragment-packed weights: 16 contiguous halves per lane (32B aligned).
__device__ __forceinline__ v16h load_b_frag(const _Float16* p) {
  const v8h lo = *(const v8h*)(p);
  const v8h hi = *(const v8h*)(p + 8);
  return __builtin_shufflevector(lo, hi, 0, 1, 2, 3, 4, 5, 6, 7,
                                 8, 9, 10, 11, 12, 13, 14, 15);
}

// One layer: (32xDIMP f16 in LDS) @ packed (DIMPxDIMP) + bias, relu -> out tile.
__device__ __forceinline__ void mlp_layer(const _Float16 (*in)[DIMP],
                                          _Float16 (*out)[DIMP],
                                          const _Float16* __restrict__ Wp,
                                          const float* __restrict__ sbias,
                                          int lane, int wv) {
  const int ksel = lane >> 4;
  const int ln15 = lane & 15;
#pragma unroll
  for (int tt = 0; tt < 3; ++tt) {          // 12 tiles / 4 waves
    const int t = wv + 4 * tt;
    const int rt = t / 6, ct = t % 6;
    F32x8 acc; acc.v = v8f{};
    const _Float16* rowp = in[rt * 16 + ln15];
#pragma unroll
    for (int kc = 0; kc < 3; ++kc) {
      v16h a  = load_a_frag(rowp, kc, ksel);
      v16h bm = load_b_frag(Wp + ((size_t)(kc * 6 + ct) * 32 + lane) * 16);
      acc.v = __builtin_amdgcn_wmma_f32_16x16x32_f16(
          false, a, false, bm, (short)0, acc.v, false, false);
    }
#pragma unroll
    for (int vi = 0; vi < 8; ++vi) {
      int m = rt * 16 + vi + 8 * ksel;
      int n = ct * 16 + ln15;
      out[m][n] = (_Float16)fmaxf(acc.e[vi] + sbias[n], 0.f);
    }
  }
}

// ---------------- fused per-center edge MLP (2 layers) + masked max over K ----
__global__ void edge_mlp_kernel(const float* __restrict__ x,
                                const float* __restrict__ pos,
                                const float* __restrict__ norm,
                                const float* __restrict__ pos_s,
                                const float* __restrict__ norm_s,
                                const int* __restrict__ nbr,
                                const float* __restrict__ b1,
                                const float* __restrict__ b2,
                                const _Float16* __restrict__ W1p,
                                const _Float16* __restrict__ W2p,
                                _Float16* __restrict__ aggh) {
  const int c   = blockIdx.x;              // center id
  const int b   = c / MM;
  const int tid = threadIdx.x;             // 128 threads = 4 waves
  const int lane = tid & 31;
  const int wv   = tid >> 5;

  __shared__ alignas(32) _Float16 bufA[KK][DIMP];
  __shared__ alignas(32) _Float16 bufB[KK][DIMP];
  __shared__ float    sppf[KK][4];
  __shared__ int      snbr[KK];
  __shared__ float    sb1[DIMP], sb2[DIMP];

  if (tid < KK) {
    int j = nbr[(size_t)c * KK + tid];
    snbr[tid] = j;
    if (j >= 0) __builtin_prefetch(x + ((size_t)b * NN + j) * NINF, 0, 3);
  }
  if (tid < DIMP) {
    sb1[tid] = (tid < DIMF) ? b1[tid] : 0.f;
    sb2[tid] = (tid < DIMF) ? b2[tid] : 0.f;
  }
  __syncthreads();

  // point-pair features per neighbor
  if (tid < KK) {
    int j = snbr[tid];
    float p0 = 0.f, p1 = 0.f, p2 = 0.f, p3 = 0.f;
    if (j >= 0) {
      const float* pj = pos + ((size_t)b * NN + j) * 3;
      const float* nj = norm + ((size_t)b * NN + j) * 3;
      const float ix = pos_s[(size_t)c * 3 + 0];
      const float iy = pos_s[(size_t)c * 3 + 1];
      const float iz = pos_s[(size_t)c * 3 + 2];
      const float nix = norm_s[(size_t)c * 3 + 0];
      const float niy = norm_s[(size_t)c * 3 + 1];
      const float niz = norm_s[(size_t)c * 3 + 2];
      const float dx = pj[0] - ix, dy = pj[1] - iy, dz = pj[2] - iz;
      p0 = sqrtf(dx * dx + dy * dy + dz * dz);
      auto ang = [](float ax, float ay, float az,
                    float bx, float by, float bz) -> float {
        float cx_ = ay * bz - az * by;
        float cy_ = az * bx - ax * bz;
        float cz_ = ax * by - ay * bx;
        float cn = sqrtf(cx_ * cx_ + cy_ * cy_ + cz_ * cz_);
        float dt = ax * bx + ay * by + az * bz;
        return atan2f(cn, dt);
      };
      p1 = ang(nix, niy, niz, dx, dy, dz);
      p2 = ang(nj[0], nj[1], nj[2], dx, dy, dz);
      p3 = ang(nix, niy, niz, nj[0], nj[1], nj[2]);
    }
    sppf[tid][0] = p0; sppf[tid][1] = p1; sppf[tid][2] = p2; sppf[tid][3] = p3;
  }
  __syncthreads();

  // build msg tile: [K x DIMP] f16, invalid rows and pad cols = 0
  for (int e = tid; e < KK * DIMP; e += 128) {
    int r = e / DIMP, cc = e % DIMP;
    int j = snbr[r];
    float v = 0.f;
    if (j >= 0) {
      if (cc < NINF)      v = x[((size_t)b * NN + j) * NINF + cc];
      else if (cc < DIMF) v = sppf[r][cc - NINF];
    }
    bufA[r][cc] = (_Float16)v;
  }
  __syncthreads();

  mlp_layer(bufA, bufB, W1p, sb1, lane, wv);   // layer 1: msg -> h1
  __syncthreads();
  mlp_layer(bufB, bufA, W2p, sb2, lane, wv);   // layer 2: h1 -> h2
  __syncthreads();

  // masked max over K (self is always valid => at least one row)
  for (int cc = tid; cc < DIMP; cc += 128) {
    float mx = -3.0e38f;
    for (int k = 0; k < KK; ++k) {
      if (snbr[k] >= 0) mx = fmaxf(mx, (float)bufA[k][cc]);
    }
    aggh[(size_t)c * DIMP + cc] = (_Float16)mx;
  }
}

// ---------------- final GEMM: agg[8192x96] @ W3[96x128] + b3, relu -------------
__global__ void final_kernel(const _Float16* __restrict__ aggh,
                             const _Float16* __restrict__ W3p,
                             const float* __restrict__ b3,
                             float* __restrict__ out) {
  const int tid = threadIdx.x;             // 256 threads = 8 waves
  const int lane = tid & 31;
  const int wv = tid >> 5;
  const int tileId = blockIdx.x * 8 + wv;  // 512 row-tiles * 8 col-tiles
  const int rt = tileId >> 3;
  const int ct = tileId & 7;
  const int ksel = lane >> 4;
  const int ln15 = lane & 15;
  const _Float16* rowp = aggh + (size_t)(rt * 16 + ln15) * DIMP;

  F32x8 acc; acc.v = v8f{};
#pragma unroll
  for (int kc = 0; kc < 3; ++kc) {
    v16h a  = load_a_frag(rowp, kc, ksel);
    v16h bm = load_b_frag(W3p + ((size_t)(kc * 8 + ct) * 32 + lane) * 16);
    acc.v = __builtin_amdgcn_wmma_f32_16x16x32_f16(
        false, a, false, bm, (short)0, acc.v, false, false);
  }
#pragma unroll
  for (int vi = 0; vi < 8; ++vi) {
    int m = rt * 16 + vi + 8 * ksel;
    int n = ct * 16 + ln15;
    out[(size_t)m * NOUTF + n] = fmaxf(acc.e[vi] + b3[n], 0.f);
  }
}

// ---------------- launcher ----------------
extern "C" void kernel_launch(void* const* d_in, const int* in_sizes, int n_in,
                              void* d_out, int out_size, void* d_ws, size_t ws_size,
                              hipStream_t stream) {
  const float* x    = (const float*)d_in[0];
  const float* pos  = (const float*)d_in[1];
  const float* norm = (const float*)d_in[2];
  // d_in[3] = batch (implicit in layout; unused)
  const float* W1 = (const float*)d_in[4];
  const float* b1 = (const float*)d_in[5];
  const float* W2 = (const float*)d_in[6];
  const float* b2 = (const float*)d_in[7];
  const float* W3 = (const float*)d_in[8];
  const float* b3 = (const float*)d_in[9];

  // output tuple flattened: out [8192,128] f32 | pos_s [8192,3] f32 | idx [8192] i32
  float* out   = (float*)d_out;
  float* pos_s = out + (size_t)NCTR * NOUTF;
  int*   idx_o = (int*)(pos_s + (size_t)NCTR * 3);

  char* ws = (char*)d_ws;
  _Float16* W1p    = (_Float16*)(ws + OFF_W1P);
  _Float16* W2p    = (_Float16*)(ws + OFF_W2P);
  _Float16* W3p    = (_Float16*)(ws + OFF_W3P);
  float*    norm_s = (float*)(ws + OFF_NRMS);
  int*      nbrp   = (int*)(ws + OFF_NBR);
  _Float16* aggh   = (_Float16*)(ws + OFF_AGG);

  prep_weights_kernel<<<(W3P_ELEMS + 255) / 256, 256, 0, stream>>>(
      W1, W2, W3, W1p, W2p, W3p);
  fps_kernel<<<BB, 256, 0, stream>>>(pos, norm, pos_s, norm_s, idx_o);
  nbr_kernel<<<NCTR, 256, 0, stream>>>(pos, pos_s, nbrp);
  edge_mlp_kernel<<<NCTR, 128, 0, stream>>>(
      x, pos, norm, pos_s, norm_s, nbrp, b1, b2, W1p, W2p, aggh);
  final_kernel<<<512, 256, 0, stream>>>(aggh, W3p, b3, out);
}
